// CausalSelfAttention_61521111548545
// MI455X (gfx1250) — matrix operand order repaired and checked by reading, not compiled, not run on hardware
//
#include <hip/hip_runtime.h>
#include <hip/hip_bf16.h>

typedef __attribute__((ext_vector_type(16))) __bf16 v16bf;
typedef __attribute__((ext_vector_type(8)))  __bf16 v8bf;
typedef __attribute__((ext_vector_type(8)))  float  v8f;
typedef __attribute__((ext_vector_type(4)))  float  v4f;

static constexpr int Bb = 64;
static constexpr int Tt = 512;
static constexpr int Cc = 384;
static constexpr int Hh = 6;
static constexpr int Dd = 64;
static constexpr int BT = Bb * Tt;        // 32768 rows

static __device__ inline v8f wmma_bf16(v16bf a, v16bf b, v8f c) {
  // D = A(16x32 bf16) * B(32x16 bf16) + C(16x16 f32)
  return __builtin_amdgcn_wmma_f32_16x16x32_bf16(false, a, false, b, (short)0, c,
                                                 false, false);
}

// CDNA5 async copy: global -> LDS, 16 bytes, tracked by ASYNCcnt.
// vdst = LDS byte offset (low 32 bits of generic shared pointer), vaddr = global addr.
static __device__ inline void async_copy_b128(void* lds_ptr, const void* gptr) {
  unsigned int lds_off = (unsigned int)(unsigned long long)lds_ptr;
  asm volatile("global_load_async_to_lds_b128 %0, %1, off"
               :: "v"(lds_off), "v"(gptr)
               : "memory");
}
static __device__ inline void wait_asynccnt0() {
  asm volatile("s_wait_asynccnt 0" ::: "memory");
}

// ---------------------------------------------------------------------------
// Kernel 1: f32 -> bf16 conversion (vectorized x4)
// ---------------------------------------------------------------------------
__global__ void __launch_bounds__(256)
cvt_f32_bf16_kernel(const float* __restrict__ in, __bf16* __restrict__ out, int n4) {
  int i = blockIdx.x * blockDim.x + threadIdx.x;
  if (i >= n4) return;
  v4f v = *(const v4f*)(in + (size_t)i * 4);
  __bf16* o = out + (size_t)i * 4;
  o[0] = (__bf16)v[0]; o[1] = (__bf16)v[1]; o[2] = (__bf16)v[2]; o[3] = (__bf16)v[3];
}

// ---------------------------------------------------------------------------
// Kernel 2: tiled bf16 WMMA GEMM.  C[M,N] = A[M,K](bf16) * B[K,N](f32->bf16)
// Block tile 128x128, K-tile 32, 256 threads = 8 waves (2x4), wave tile 64x32.
// M,N multiples of 128; K multiple of 32 (holds for all our shapes).
// A tile staged via GLOBAL_LOAD_ASYNC_TO_LDS_B128 (overlaps B convert/transpose).
// ---------------------------------------------------------------------------
template <bool OUT_F32>
__global__ void __launch_bounds__(256)
gemm_bf16_kernel(const __bf16* __restrict__ A, const float* __restrict__ Bw,
                 void* __restrict__ Cout, int N, int K) {
  __shared__ __bf16 Alds[128 * 40];   // row-major [m][k], stride 40 (80B = 5*16B)
  __shared__ __bf16 Blds[128 * 40];   // transposed [n][k], stride 40

  const int tid  = threadIdx.x;
  const int wave = tid >> 5;
  const int lane = tid & 31;
  const int lm   = lane & 15;
  const int lh   = lane >> 4;
  const int wm   = wave >> 2;         // 0..1  -> 64-row strip
  const int wn   = wave & 3;          // 0..3  -> 32-col strip
  const int m0   = blockIdx.x * 128;
  const int n0   = blockIdx.y * 128;

  v8f acc[4][2];
#pragma unroll
  for (int mf = 0; mf < 4; ++mf)
#pragma unroll
    for (int nf = 0; nf < 2; ++nf)
#pragma unroll
      for (int r = 0; r < 8; ++r) acc[mf][nf][r] = 0.0f;

  for (int k0 = 0; k0 < K; k0 += 32) {
    __syncthreads();
    // ---- async-stage A tile: 128x32 bf16, 512 chunks of 8 elements (16B each)
#pragma unroll
    for (int c = 0; c < 2; ++c) {
      int chunk = c * 256 + tid;
      int row = chunk >> 2;
      int col = (chunk & 3) * 8;
      async_copy_b128(Alds + row * 40 + col,
                      A + (size_t)(m0 + row) * K + k0 + col);
    }
    // ---- load B tile: 32x128 f32 -> bf16 transposed into Blds[n][k]
    //      (VALU convert work overlaps the async DMA above)
#pragma unroll
    for (int c = 0; c < 2; ++c) {
      int chunk = c * 256 + tid;
      int krow  = chunk >> 4;          // 16 chunks (of 8 floats) per k-row
      int ncol  = (chunk & 15) * 8;
      const float* src = Bw + (size_t)(k0 + krow) * N + n0 + ncol;
      v4f f0 = *(const v4f*)src;
      v4f f1 = *(const v4f*)(src + 4);
#pragma unroll
      for (int j = 0; j < 4; ++j) Blds[(ncol + j) * 40 + krow]     = (__bf16)f0[j];
#pragma unroll
      for (int j = 0; j < 4; ++j) Blds[(ncol + 4 + j) * 40 + krow] = (__bf16)f1[j];
    }
    wait_asynccnt0();
    __syncthreads();

    // ---- build fragments (ISA 7.12.2 layouts) and issue WMMAs
    v16bf afr[4], bfr[2];
#pragma unroll
    for (int mf = 0; mf < 4; ++mf) {
      const __bf16* p = Alds + (wm * 64 + mf * 16 + lm) * 40;
      v8bf lo = *(const v8bf*)(p + lh * 8);
      v8bf hi = *(const v8bf*)(p + 16 + lh * 8);
#pragma unroll
      for (int e = 0; e < 8; ++e) { afr[mf][e] = lo[e]; afr[mf][8 + e] = hi[e]; }
    }
#pragma unroll
    for (int nf = 0; nf < 2; ++nf) {
      const __bf16* p = Blds + (wn * 32 + nf * 16 + lm) * 40;
      v8bf lo = *(const v8bf*)(p + lh * 8);
      v8bf hi = *(const v8bf*)(p + 16 + lh * 8);
#pragma unroll
      for (int e = 0; e < 8; ++e) { bfr[nf][e] = lo[e]; bfr[nf][8 + e] = hi[e]; }
    }
#pragma unroll
    for (int mf = 0; mf < 4; ++mf)
#pragma unroll
      for (int nf = 0; nf < 2; ++nf)
        acc[mf][nf] = wmma_bf16(afr[mf], bfr[nf], acc[mf][nf]);
  }

  // ---- store C fragments: VGPR r -> row r + 8*lh, lanes(%16) -> column.
  //      32-bit incremental indexing (max index < 2^31), branch-free.
  float*  cf = (float*)Cout;
  __bf16* cb = (__bf16*)Cout;
  const unsigned colbase = (unsigned)(n0 + wn * 32 + lm);
#pragma unroll
  for (int mf = 0; mf < 4; ++mf) {
    unsigned idx = (unsigned)(m0 + wm * 64 + mf * 16 + lh * 8) * (unsigned)N + colbase;
#pragma unroll
    for (int r = 0; r < 8; ++r) {
#pragma unroll
      for (int nf = 0; nf < 2; ++nf) {
        if (OUT_F32) cf[idx + nf * 16] = acc[mf][nf][r];
        else         cb[idx + nf * 16] = (__bf16)acc[mf][nf][r];
      }
      idx += (unsigned)N;
    }
  }
}

// ---------------------------------------------------------------------------
// Kernel 3: per-head RMSNorm + RoPE, in place on bf16 qkv [BT, 1152].
// grid = B*T*H blocks of 64 threads; wave 0 -> q, wave 1 -> k.
// Lane i owns the rotate-half pair (d=i, d=i+32).
// ---------------------------------------------------------------------------
__global__ void __launch_bounds__(64)
rmsnorm_rope_kernel(__bf16* __restrict__ qkv,
                    const float* __restrict__ qw, const float* __restrict__ kw) {
  const int bth  = blockIdx.x;
  const int h    = bth % Hh;
  const int bt   = bth / Hh;
  const int t    = bt % Tt;
  const int wave = threadIdx.x >> 5;
  const int lane = threadIdx.x & 31;

  __bf16* base = qkv + (size_t)bt * (3 * Cc) + wave * Cc + h * Dd;
  const float* w = wave ? kw : qw;

  float v1 = (float)base[lane];
  float v2 = (float)base[lane + 32];
  float ss = v1 * v1 + v2 * v2;
#pragma unroll
  for (int m = 1; m < 32; m <<= 1) ss += __shfl_xor(ss, m, 32);
  float r  = rsqrtf(ss * (1.0f / 64.0f) + 1e-6f);
  float n1 = v1 * r * w[lane];
  float n2 = v2 * r * w[lane + 32];

  float invf = __powf(10000.0f, -(float)(2 * lane) / 64.0f);
  float ang  = (float)t * invf;
  float cs = __cosf(ang), sn = __sinf(ang);
  base[lane]      = (__bf16)(n1 * cs - n2 * sn);
  base[lane + 32] = (__bf16)(n2 * cs + n1 * sn);
}

// ---------------------------------------------------------------------------
// Kernel 4: causal flash attention.  grid = (T/64, B*H), 128 threads = 4 waves.
// Each wave owns 16 query rows; KV tiled 64 at a time through LDS.
// K tile staged with async-to-LDS DMA; V transposed synchronously (overlaps).
// ---------------------------------------------------------------------------
__global__ void __launch_bounds__(128)
flash_attn_kernel(const __bf16* __restrict__ qkv, __bf16* __restrict__ y) {
  __shared__ __bf16 Klds[64 * 72];          // [kv][d], stride 72 (144B = 9*16B)
  __shared__ __bf16 Vlds[64 * 72];          // transposed [d][kv]
  __shared__ __bf16 Plds[4 * 16 * 72];      // per-wave P strip [16][64]

  const int bh = blockIdx.y;
  const int b  = bh / Hh;
  const int h  = bh % Hh;
  const int qt = blockIdx.x;
  const int tid  = threadIdx.x;
  const int wave = tid >> 5;
  const int lane = tid & 31;
  const int lm   = lane & 15;
  const int lh   = lane >> 4;
  const int qr0  = qt * 64 + wave * 16;

  // ---- load Q A-fragments once (D=64 -> 2 K-slices of 32)
  v16bf aq[2];
  {
    const __bf16* qp = qkv + (size_t)(b * Tt + qr0 + lm) * (3 * Cc) + h * Dd;
#pragma unroll
    for (int kt = 0; kt < 2; ++kt) {
      v8bf lo = *(const v8bf*)(qp + kt * 32 + lh * 8);
      v8bf hi = *(const v8bf*)(qp + kt * 32 + 16 + lh * 8);
#pragma unroll
      for (int e = 0; e < 8; ++e) { aq[kt][e] = lo[e]; aq[kt][8 + e] = hi[e]; }
    }
  }

  float mrow[8], lrow[8];
  v8f of[4];
#pragma unroll
  for (int r = 0; r < 8; ++r) { mrow[r] = -3.0e38f; lrow[r] = 0.0f; }
#pragma unroll
  for (int nf = 0; nf < 4; ++nf)
#pragma unroll
    for (int r = 0; r < 8; ++r) of[nf][r] = 0.0f;

  const float scale = 0.125f;  // 1/sqrt(64)

  for (int kvt = 0; kvt <= qt; ++kvt) {
    const int kv0 = kvt * 64;
    __syncthreads();
    // ---- stage K (async DMA, row-major) and V (transposed, sync) tiles
    {
      int row = tid >> 1;
      int c0  = (tid & 1) * 32;
      const __bf16* kp = qkv + (size_t)(b * Tt + kv0 + row) * (3 * Cc) + Cc + h * Dd + c0;
      const __bf16* vp = kp + Cc;  // v is 384 past k
#pragma unroll
      for (int j = 0; j < 4; ++j)
        async_copy_b128(Klds + row * 72 + c0 + j * 8, kp + j * 8);
#pragma unroll
      for (int j = 0; j < 4; ++j) {
        v8bf vv8 = *(const v8bf*)(vp + j * 8);
#pragma unroll
        for (int e = 0; e < 8; ++e) Vlds[(c0 + j * 8 + e) * 72 + row] = vv8[e];
      }
    }
    wait_asynccnt0();
    __syncthreads();

    // ---- S = Q * K^T  (4 n-frags of 16 kv columns, K-dim = D = 2x32)
    v8f sf[4];
#pragma unroll
    for (int nf = 0; nf < 4; ++nf) {
      v8f c;
#pragma unroll
      for (int r = 0; r < 8; ++r) c[r] = 0.0f;
#pragma unroll
      for (int kt = 0; kt < 2; ++kt) {
        const __bf16* p = Klds + (nf * 16 + lm) * 72 + kt * 32;
        v8bf lo = *(const v8bf*)(p + lh * 8);
        v8bf hi = *(const v8bf*)(p + 16 + lh * 8);
        v16bf bk;
#pragma unroll
        for (int e = 0; e < 8; ++e) { bk[e] = lo[e]; bk[8 + e] = hi[e]; }
        c = wmma_bf16(aq[kt], bk, c);
      }
      sf[nf] = c;
    }

    // ---- scale + causal mask (only diagonal tile needs it)
    const bool diag = (kvt == qt);
#pragma unroll
    for (int nf = 0; nf < 4; ++nf)
#pragma unroll
      for (int r = 0; r < 8; ++r) {
        float val = sf[nf][r] * scale;
        if (diag) {
          int col = kv0 + nf * 16 + lm;
          int row = qr0 + r + lh * 8;
          if (col > row) val = -3.0e38f;
        }
        sf[nf][r] = val;
      }

    // ---- online softmax: row stats live redundantly in each 16-lane half
    float mnew[8], sc[8];
#pragma unroll
    for (int r = 0; r < 8; ++r) {
      float mx = fmaxf(fmaxf(sf[0][r], sf[1][r]), fmaxf(sf[2][r], sf[3][r]));
#pragma unroll
      for (int m = 1; m < 16; m <<= 1) mx = fmaxf(mx, __shfl_xor(mx, m, 32));
      mnew[r] = fmaxf(mrow[r], mx);
      sc[r]   = __expf(mrow[r] - mnew[r]);
      mrow[r] = mnew[r];
    }

    float rsum[8];
#pragma unroll
    for (int r = 0; r < 8; ++r) rsum[r] = 0.0f;
    __bf16* pw = Plds + wave * 16 * 72;
#pragma unroll
    for (int nf = 0; nf < 4; ++nf)
#pragma unroll
      for (int r = 0; r < 8; ++r) {
        float p = __expf(sf[nf][r] - mnew[r]);
        rsum[r] += p;
        pw[(r + lh * 8) * 72 + nf * 16 + lm] = (__bf16)p;
      }
#pragma unroll
    for (int r = 0; r < 8; ++r) {
      float s = rsum[r];
#pragma unroll
      for (int m = 1; m < 16; m <<= 1) s += __shfl_xor(s, m, 32);
      lrow[r] = lrow[r] * sc[r] + s;
    }
#pragma unroll
    for (int nf = 0; nf < 4; ++nf)
#pragma unroll
      for (int r = 0; r < 8; ++r) of[nf][r] *= sc[r];

    // ---- O += P * V   (P reloaded from LDS as A-fragments; same-wave region)
#pragma unroll
    for (int kt = 0; kt < 2; ++kt) {
      const __bf16* p = pw + lm * 72;
      v8bf lo = *(const v8bf*)(p + kt * 32 + lh * 8);
      v8bf hi = *(const v8bf*)(p + kt * 32 + 16 + lh * 8);
      v16bf ap;
#pragma unroll
      for (int e = 0; e < 8; ++e) { ap[e] = lo[e]; ap[8 + e] = hi[e]; }
#pragma unroll
      for (int nf = 0; nf < 4; ++nf) {
        const __bf16* vp2 = Vlds + (nf * 16 + lm) * 72 + kt * 32;
        v8bf blo = *(const v8bf*)(vp2 + lh * 8);
        v8bf bhi = *(const v8bf*)(vp2 + 16 + lh * 8);
        v16bf bv;
#pragma unroll
        for (int e = 0; e < 8; ++e) { bv[e] = blo[e]; bv[8 + e] = bhi[e]; }
        of[nf] = wmma_bf16(ap, bv, of[nf]);
      }
    }
  }

  // ---- normalize and store y in [B,T,H*D] layout (bf16) for the proj GEMM
  const unsigned colbase = (unsigned)(h * Dd + lm);
#pragma unroll
  for (int nf = 0; nf < 4; ++nf) {
    unsigned idx = (unsigned)(b * Tt + qr0 + lh * 8) * (unsigned)Cc + colbase + nf * 16;
#pragma unroll
    for (int r = 0; r < 8; ++r) {
      y[idx] = (__bf16)(of[nf][r] / lrow[r]);
      idx += (unsigned)Cc;
    }
  }
}

// ---------------------------------------------------------------------------
// Host-side launcher
// ---------------------------------------------------------------------------
extern "C" void kernel_launch(void* const* d_in, const int* in_sizes, int n_in,
                              void* d_out, int out_size, void* d_ws, size_t ws_size,
                              hipStream_t stream) {
  (void)in_sizes; (void)n_in; (void)out_size; (void)ws_size;
  const float* x      = (const float*)d_in[0];   // [B,T,C]
  const float* w_qkv  = (const float*)d_in[1];   // [C, 3C]
  const float* w_proj = (const float*)d_in[2];   // [C, C]
  const float* q_w    = (const float*)d_in[3];   // [D]
  const float* k_w    = (const float*)d_in[4];   // [D]
  float* out = (float*)d_out;                    // [B,T,C] f32

  // workspace layout (bytes)
  char* ws = (char*)d_ws;
  __bf16* x_bf   = (__bf16*)(ws);                                  // BT*C   bf16
  __bf16* qkv_bf = (__bf16*)(ws + (size_t)BT * Cc * 2);            // BT*3C  bf16
  __bf16* y_bf   = (__bf16*)(ws + (size_t)BT * Cc * 2 + (size_t)BT * 3 * Cc * 2);

  // 1) x -> bf16
  {
    int n4 = (BT * Cc) / 4;
    cvt_f32_bf16_kernel<<<(n4 + 255) / 256, 256, 0, stream>>>(x, x_bf, n4);
  }
  // 2) qkv = x @ w_qkv   (32768 x 1152, K=384) -> bf16
  {
    dim3 grid(BT / 128, (3 * Cc) / 128);
    gemm_bf16_kernel<false><<<grid, 256, 0, stream>>>(x_bf, w_qkv, (void*)qkv_bf,
                                                      3 * Cc, Cc);
  }
  // 3) RMSNorm + RoPE in place on q,k
  {
    rmsnorm_rope_kernel<<<BT * Hh, 64, 0, stream>>>(qkv_bf, q_w, k_w);
  }
  // 4) causal flash attention -> y_bf [B,T,C]
  {
    dim3 grid(Tt / 64, Bb * Hh);
    flash_attn_kernel<<<grid, 128, 0, stream>>>(qkv_bf, y_bf);
  }
  // 5) out = y @ w_proj  (32768 x 384, K=384) -> f32
  {
    dim3 grid(BT / 128, Cc / 128);
    gemm_bf16_kernel<true><<<grid, 256, 0, stream>>>(y_bf, w_proj, (void*)out,
                                                     Cc, Cc);
  }
}
